// CEPPairStack_43130061586676
// MI455X (gfx1250) — matrix operand order repaired and checked by reading, not compiled, not run on hardware
//
#include <hip/hip_runtime.h>
#include <hip/hip_bf16.h>

#define USE_ASYNC 1   // global_load_async_to_lds_b128 staging (ASYNCcnt)
#define USE_TDM   1   // tensor_load_to_lds for weight tiles (TENSORcnt)

// ---------------------------------------------------------------------------
// Types / helpers
// ---------------------------------------------------------------------------
typedef __attribute__((ext_vector_type(16))) __bf16 bf16x16;
typedef __attribute__((ext_vector_type(8)))  __bf16 bf16x8;
typedef __attribute__((ext_vector_type(8)))  float  f32x8;
typedef __attribute__((ext_vector_type(4)))  unsigned u32x4;
typedef __attribute__((ext_vector_type(8)))  unsigned u32x8;

static __device__ __forceinline__ __bf16 f2bf(float f) {
  unsigned u = __builtin_bit_cast(unsigned, f);
  unsigned r = (u + 0x7FFFu + ((u >> 16) & 1u)) >> 16;
  unsigned short s = (unsigned short)r;
  return __builtin_bit_cast(__bf16, s);
}
static __device__ __forceinline__ float bf2f(__bf16 b) {
  unsigned short s = __builtin_bit_cast(unsigned short, b);
  unsigned u = ((unsigned)s) << 16;
  return __builtin_bit_cast(float, u);
}
static __device__ __forceinline__ float sigmoidf(float x) {
  return 1.0f / (1.0f + __expf(-x));
}

// ---- CDNA5 async / tensor data movement ----------------------------------
template <int N>
static __device__ __forceinline__ void wait_asynccnt() {
  asm volatile("s_wait_asynccnt %0" ::"i"(N) : "memory");
}
template <int N>
static __device__ __forceinline__ void wait_tensorcnt() {
  asm volatile("s_wait_tensorcnt %0" ::"i"(N) : "memory");
}
// Async copy 16B from global to LDS (per-lane addresses). LDS address is the
// low 32 bits of the generic pointer (aperture holds the high half).
static __device__ __forceinline__ void async_load_b128(void* ldsPtr, const void* g) {
  unsigned lds = (unsigned)(size_t)ldsPtr;
  unsigned long long ga = (unsigned long long)(size_t)g;
  asm volatile("global_load_async_to_lds_b128 %0, %1, off" ::"v"(lds), "v"(ga)
               : "memory");
}
// TDM: load a 2D tile (tileW x tileH elements of 2 bytes) from a row-major
// tensor with row stride strideElems into LDS at ldsPtr, inserting
// 16B of padding after every 64B row chunk (pad_interval=16 DW, pad_amount=4 DW)
// so the LDS layout matches a 40-element padded row of a 32-element tile.
static __device__ __forceinline__ void tdm_load_2d_bf16(
    void* ldsPtr, const void* gaddr, unsigned dim0Remain, unsigned dim1Remain,
    unsigned tileW, unsigned tileH, unsigned strideElems) {
  unsigned long long ga = (unsigned long long)(size_t)gaddr;
  u32x4 g0;
  g0[0] = 1u;  // count=1, user mode, no gather
  g0[1] = (unsigned)(size_t)ldsPtr;
  g0[2] = (unsigned)ga;                                        // global addr [31:0]
  g0[3] = (unsigned)((ga >> 32) & 0x1FFFFFFu) | (2u << 30);    // addr[56:32] | type=2
  u32x8 g1;
  g1[0] = (1u << 16)        // data_size = 1 -> 2 bytes
        | (1u << 20)        // pad_enable
        | (3u << 22)        // pad_interval: 16 DWORDs (64B) before padding
        | (3u << 25);       // pad_amount: 4 DWORDs (16B)
  g1[1] = (dim0Remain & 0xFFFFu) << 16;                        // tensor_dim0 lo16
  g1[2] = (dim0Remain >> 16) | ((dim1Remain & 0xFFFFu) << 16); // dim0 hi16 | dim1 lo16
  g1[3] = (dim1Remain >> 16) | (tileW << 16);                  // dim1 hi16 | tile_dim0
  g1[4] = tileH & 0xFFFFu;                                     // tile_dim1 (tile_dim2=0)
  g1[5] = strideElems;                                         // tensor_dim0_stride lo32
  g1[6] = 0u;                                                  // stride hi16 | dim1_stride lo16
  g1[7] = 0u;
  asm volatile("tensor_load_to_lds %0, %1" ::"s"(g0), "s"(g1) : "memory");
}

union FragU { bf16x16 v; bf16x8 h[2]; };

// A fragment (16xK=32, bf16): ISA layout: element e -> K = 16*(e>>3) + 8*half + (e&7)
static __device__ __forceinline__ bf16x16 load_frag_a(const __bf16* lds, int ld, int mBase,
                                                      int kBase, int lane) {
  int half = lane >> 4, mr = lane & 15;
  const __bf16* p = lds + (mBase + mr) * ld + kBase;
  FragU f;
  f.h[0] = *(const bf16x8*)(p + 8 * half);
  f.h[1] = *(const bf16x8*)(p + 16 + 8 * half);
  return f.v;
}
// B fragment (K=32 x 16, bf16), lds holds rows indexed by n (N-major), K contiguous:
// element e -> K = 16*(e>>3) + 8*half + (e&7)
static __device__ __forceinline__ bf16x16 load_frag_b(const __bf16* lds, int ld, int nBase,
                                                      int kBase, int lane) {
  int half = lane >> 4, nr = lane & 15;
  const __bf16* p = lds + (nBase + nr) * ld + kBase;
  FragU f;
  f.h[0] = *(const bf16x8*)(p + 8 * half);
  f.h[1] = *(const bf16x8*)(p + 16 + 8 * half);
  return f.v;
}
static __device__ __forceinline__ f32x8 wmma_bf16(bf16x16 a, bf16x16 b, f32x8 c) {
  return __builtin_amdgcn_wmma_f32_16x16x32_bf16(false, a, false, b, (short)0, c, false, false);
}

#define GF_RELU     1
#define GF_BF16OUT  2
#define GF_MASKPRE  4
#define GF_MASKPOST 8

// ---------------------------------------------------------------------------
// LayerNorm over 128 channels, one wave per row.
// ---------------------------------------------------------------------------
__global__ __launch_bounds__(256) void ln_kernel(const float* __restrict__ x,
                                                 const float* __restrict__ g,
                                                 const float* __restrict__ be,
                                                 __bf16* __restrict__ obf,
                                                 float* __restrict__ of32,
                                                 const float* __restrict__ maskRow,
                                                 int nRows) {
  int wid = threadIdx.x >> 5, lane = threadIdx.x & 31;
  int row = blockIdx.x * 8 + wid;
  if (row >= nRows) return;
  const float* xr = x + (size_t)row * 128;
  float4 v = *(const float4*)(xr + lane * 4);
  float s = v.x + v.y + v.z + v.w;
#pragma unroll
  for (int m = 16; m >= 1; m >>= 1) s += __shfl_xor(s, m, 32);
  float mu = s * (1.0f / 128.0f);
  float dx = v.x - mu, dy = v.y - mu, dz = v.z - mu, dw = v.w - mu;
  float q = dx * dx + dy * dy + dz * dz + dw * dw;
#pragma unroll
  for (int m = 16; m >= 1; m >>= 1) q += __shfl_xor(q, m, 32);
  float rs = rsqrtf(q * (1.0f / 128.0f) + 1e-5f);
  float4 gg = *(const float4*)(g + lane * 4);
  float4 bb = *(const float4*)(be + lane * 4);
  float o0 = dx * rs * gg.x + bb.x;
  float o1 = dy * rs * gg.y + bb.y;
  float o2 = dz * rs * gg.z + bb.z;
  float o3 = dw * rs * gg.w + bb.w;
  if (obf) {
    __bf16* op = obf + (size_t)row * 128 + lane * 4;
    op[0] = f2bf(o0); op[1] = f2bf(o1); op[2] = f2bf(o2); op[3] = f2bf(o3);
  }
  if (of32) {
    float mk = maskRow ? maskRow[row] : 1.0f;
    float4 ov = make_float4(o0 * mk, o1 * mk, o2 * mk, o3 * mk);
    *(float4*)(of32 + (size_t)row * 128 + lane * 4) = ov;
  }
}

// ---------------------------------------------------------------------------
// Generic bf16 WMMA GEMM: Y[M,N] = X[M,K] @ W[K,N] (WT stored [N][K]).
// Block tile 128x64, BK=32, 256 threads (8 waves), each wave -> 32x32.
// Double-buffered LDS: X tile via async-to-LDS loads (ASYNCcnt), W tile via
// Tensor Data Mover (TENSORcnt) issued by wave 0.
// ---------------------------------------------------------------------------
__global__ __launch_bounds__(256) void gemm_kernel(
    const __bf16* __restrict__ X, int ldX, const __bf16* __restrict__ WT, int K,
    float* __restrict__ Yf, __bf16* __restrict__ Ybf, int ldY,
    const float* __restrict__ bias, const float* __restrict__ gate,
    const float* __restrict__ maskRow, const float* __restrict__ res, int flags) {
  __shared__ __attribute__((aligned(16))) __bf16 Xs[2][128 * 40];
  __shared__ __attribute__((aligned(16))) __bf16 Ws[2][64 * 40];
  int m0 = blockIdx.x * 128, n0 = blockIdx.y * 64;
  int tid = threadIdx.x, wid = tid >> 5, lane = tid & 31;
  int wm = wid >> 1, wn = wid & 1;
  int nRowsW = (int)gridDim.y * 64;
  // per-thread X staging chunks: 2 x 16B per k-step
  int xr0 = tid >> 2, xc = (tid & 3) * 8;
  int xr1 = xr0 + 64;

  auto stageX = [&](int buf, int k0) {
    const __bf16* s0 = X + (size_t)(m0 + xr0) * ldX + k0 + xc;
    const __bf16* s1 = X + (size_t)(m0 + xr1) * ldX + k0 + xc;
#if USE_ASYNC
    async_load_b128(&Xs[buf][xr0 * 40 + xc], s0);
    async_load_b128(&Xs[buf][xr1 * 40 + xc], s1);
#else
    *(bf16x8*)&Xs[buf][xr0 * 40 + xc] = *(const bf16x8*)s0;
    *(bf16x8*)&Xs[buf][xr1 * 40 + xc] = *(const bf16x8*)s1;
#endif
  };
  auto stageW = [&](int buf, int k0) {
#if USE_TDM
    if (wid == 0)
      tdm_load_2d_bf16(&Ws[buf][0], WT + (size_t)n0 * K + k0,
                       (unsigned)(K - k0), (unsigned)(nRowsW - n0),
                       32u, 64u, (unsigned)K);
#else
    int r = tid >> 2, cc = (tid & 3) * 8;
    const __bf16* src = WT + (size_t)(n0 + r) * K + k0 + cc;
#if USE_ASYNC
    async_load_b128(&Ws[buf][r * 40 + cc], src);
#else
    *(bf16x8*)&Ws[buf][r * 40 + cc] = *(const bf16x8*)src;
#endif
#endif
  };

  f32x8 acc[2][2] = {};
  stageX(0, 0);
  stageW(0, 0);
  int nsteps = K >> 5;
  for (int s = 0; s < nsteps; s++) {
    int cur = s & 1, nxt = cur ^ 1;
    if (s + 1 < nsteps) {
      stageX(nxt, (s + 1) * 32);
      stageW(nxt, (s + 1) * 32);
#if USE_ASYNC
      wait_asynccnt<2>();   // retire previous batch (in-order), keep next in flight
#endif
#if USE_TDM
      if (wid == 0) wait_tensorcnt<1>();
#endif
    } else {
#if USE_ASYNC
      wait_asynccnt<0>();
#endif
#if USE_TDM
      if (wid == 0) wait_tensorcnt<0>();
#endif
    }
    __syncthreads();
    bf16x16 a0 = load_frag_a(Xs[cur], 40, wm * 32, 0, lane);
    bf16x16 a1 = load_frag_a(Xs[cur], 40, wm * 32 + 16, 0, lane);
    bf16x16 b0 = load_frag_b(Ws[cur], 40, wn * 32, 0, lane);
    bf16x16 b1 = load_frag_b(Ws[cur], 40, wn * 32 + 16, 0, lane);
    acc[0][0] = wmma_bf16(a0, b0, acc[0][0]);
    acc[0][1] = wmma_bf16(a0, b1, acc[0][1]);
    acc[1][0] = wmma_bf16(a1, b0, acc[1][0]);
    acc[1][1] = wmma_bf16(a1, b1, acc[1][1]);
    __syncthreads();
  }
  int half = lane >> 4, nr = lane & 15;
#pragma unroll
  for (int ai = 0; ai < 2; ai++)
#pragma unroll
    for (int bi = 0; bi < 2; bi++) {
      int mBase = m0 + wm * 32 + ai * 16 + 8 * half;
      int n = n0 + wn * 32 + bi * 16 + nr;
      float bv = bias ? bias[n] : 0.0f;
#pragma unroll
      for (int r = 0; r < 8; r++) {
        size_t row = (size_t)(mBase + r);
        float v = acc[ai][bi][r] + bv;
        if (flags & GF_RELU) v = fmaxf(v, 0.0f);
        if (gate) v *= sigmoidf(gate[row * ldY + n]);
        float mk = maskRow ? maskRow[row] : 1.0f;
        if (flags & GF_MASKPRE) v *= mk;
        if (res) v += res[row * ldY + n];
        if (flags & GF_MASKPOST) v *= mk;
        if (flags & GF_BF16OUT) Ybf[row * ldY + n] = f2bf(v);
        else                    Yf[row * ldY + n] = v;
      }
    }
}

// ---------------------------------------------------------------------------
// Triangle attention (flash style). One block = 128 queries x 256 keys for a
// fixed (outer index t, head h). Stride params select "starting" (row) vs
// "ending" (column) attention. qkvg: bf16 [M][512] (q|k|v|g interleaved c*4+h).
// ---------------------------------------------------------------------------
__global__ __launch_bounds__(256) void attn_kernel(
    const __bf16* __restrict__ qkvg, const float* __restrict__ pb,
    const float* __restrict__ bg, const float* __restrict__ mask,
    __bf16* __restrict__ outO,
    int outerMul, int innerMul, int biasQMul, int biasKMul,
    int maskTMul, int maskKMul) {
  __shared__ __attribute__((aligned(16))) __bf16 Qs[128 * 40];
  __shared__ __attribute__((aligned(16))) __bf16 Ks[256 * 40];
  __shared__ __attribute__((aligned(16))) __bf16 Vt[32 * 264];
  __shared__ __attribute__((aligned(16))) __bf16 Pw[8 * 16 * 40];
  __shared__ float mk_s[256];
  int t = blockIdx.x;
  int h = blockIdx.y & 3, qb = blockIdx.y >> 2;
  int q0 = qb * 128;
  int tid = threadIdx.x, wid = tid >> 5, lane = tid & 31;

  if (tid < 256) mk_s[tid] = mask[(size_t)t * maskTMul + (size_t)tid * maskKMul];
  for (int idx = tid; idx < 256 * 32; idx += 256) {
    int kk = idx >> 5, c = idx & 31;
    size_t rowp = (size_t)t * outerMul + (size_t)kk * innerMul;
    Ks[kk * 40 + c] = qkvg[rowp * 512 + 128 + c * 4 + h];
    Vt[c * 264 + kk] = qkvg[rowp * 512 + 256 + c * 4 + h];
  }
  for (int idx = tid; idx < 128 * 32; idx += 256) {
    int jq = idx >> 5, c = idx & 31;
    size_t rowp = (size_t)t * outerMul + (size_t)(q0 + jq) * innerMul;
    Qs[jq * 40 + c] = qkvg[rowp * 512 + c * 4 + h];
  }
  __syncthreads();

  int half = lane >> 4, nr = lane & 15;
  bf16x16 aq = load_frag_a(Qs, 40, wid * 16, 0, lane);
  f32x8 o0 = {}, o1 = {};
  float mrun[8], lrun[8];
#pragma unroll
  for (int r = 0; r < 8; r++) { mrun[r] = -1e30f; lrun[r] = 0.0f; }
  const float scale = 0.17677669529663687f;  // 1/sqrt(32)
  __bf16* pw = &Pw[wid * 16 * 40];

  for (int k0 = 0; k0 < 256; k0 += 32) {
    f32x8 s0 = {}, s1 = {};
    s0 = wmma_bf16(aq, load_frag_b(Ks, 40, k0, 0, lane), s0);
    s1 = wmma_bf16(aq, load_frag_b(Ks, 40, k0 + 16, 0, lane), s1);
    float sv[2][8];
    float stepmax[8];
#pragma unroll
    for (int r = 0; r < 8; r++) stepmax[r] = -1e30f;
#pragma unroll
    for (int tt = 0; tt < 2; tt++) {
      int kkb = k0 + tt * 16 + nr;
      float mkv = mk_s[kkb];
#pragma unroll
      for (int r = 0; r < 8; r++) {
        int qg = q0 + wid * 16 + r + 8 * half;
        float v = (tt ? s1[r] : s0[r]) * scale +
                  pb[(size_t)(qg * biasQMul + kkb * biasKMul) * 4 + h];
        v = (v + 100.0f) * mkv - 100.0f;
        sv[tt][r] = v;
        stepmax[r] = fmaxf(stepmax[r], v);
      }
    }
#pragma unroll
    for (int r = 0; r < 8; r++) {
      float m = stepmax[r];
#pragma unroll
      for (int sft = 1; sft < 16; sft <<= 1) m = fmaxf(m, __shfl_xor(m, sft, 32));
      float newm = fmaxf(mrun[r], m);
      float alpha = __expf(mrun[r] - newm);
      o0[r] *= alpha; o1[r] *= alpha;
      lrun[r] *= alpha;
      mrun[r] = newm;
    }
#pragma unroll
    for (int tt = 0; tt < 2; tt++)
#pragma unroll
      for (int r = 0; r < 8; r++) sv[tt][r] = __expf(sv[tt][r] - mrun[r]);
#pragma unroll
    for (int r = 0; r < 8; r++) {
      float s = sv[0][r] + sv[1][r];
#pragma unroll
      for (int sft = 1; sft < 16; sft <<= 1) s += __shfl_xor(s, sft, 32);
      lrun[r] += s;
    }
    // C-layout -> A-layout via per-wave LDS buffer
#pragma unroll
    for (int tt = 0; tt < 2; tt++)
#pragma unroll
      for (int r = 0; r < 8; r++)
        pw[(r + 8 * half) * 40 + tt * 16 + nr] = f2bf(sv[tt][r]);
    bf16x16 ap = load_frag_a(pw, 40, 0, 0, lane);
    o0 = wmma_bf16(ap, load_frag_b(Vt, 264, 0, k0, lane), o0);
    o1 = wmma_bf16(ap, load_frag_b(Vt, 264, 16, k0, lane), o1);
  }
#pragma unroll
  for (int ct = 0; ct < 2; ct++)
#pragma unroll
    for (int r = 0; r < 8; r++) {
      int q = wid * 16 + r + 8 * half;
      int c = ct * 16 + nr;
      size_t rowp = (size_t)t * outerMul + (size_t)(q0 + q) * innerMul;
      float gsum = bf2f(qkvg[rowp * 512 + 384 + c * 4 + h]) + bg[c * 4 + h];
      float v = (ct ? o1[r] : o0[r]) / lrun[r] * sigmoidf(gsum);
      outO[rowp * 128 + c * 4 + h] = f2bf(v);
    }
}

// ---------------------------------------------------------------------------
// Triangle-mul einsum: per channel c, Out_c[i,j] = sum_k A[f(i,k)][c]*B[f(j,k)][c]
// with f(i,k) = i*iMul + k*kMul. Block: 64x64 output tile for one channel.
// ---------------------------------------------------------------------------
__global__ __launch_bounds__(256) void trimul_kernel(
    const __bf16* __restrict__ A, const __bf16* __restrict__ Bb,
    float* __restrict__ Out, int iMul, int kMul) {
  __shared__ __attribute__((aligned(16))) __bf16 As[64 * 40];
  __shared__ __attribute__((aligned(16))) __bf16 Bs[64 * 40];
  int ti = blockIdx.x >> 2, tj = blockIdx.x & 3;
  int c = blockIdx.y;
  int i0 = ti * 64, j0 = tj * 64;
  int tid = threadIdx.x, wid = tid >> 5, lane = tid & 31;
  f32x8 acc[2] = {};
  int t0 = wid * 2;
  for (int k0 = 0; k0 < 256; k0 += 32) {
    for (int idx = tid; idx < 64 * 32; idx += 256) {
      int il = idx >> 5, kl = idx & 31;
      As[il * 40 + kl] = A[(size_t)((i0 + il) * iMul + (k0 + kl) * kMul) * 128 + c];
      Bs[il * 40 + kl] = Bb[(size_t)((j0 + il) * iMul + (k0 + kl) * kMul) * 128 + c];
    }
    __syncthreads();
#pragma unroll
    for (int tt = 0; tt < 2; tt++) {
      int tile = t0 + tt;
      int mt = tile >> 2, nt = tile & 3;
      bf16x16 af = load_frag_a(As, 40, mt * 16, 0, lane);
      bf16x16 bfr = load_frag_b(Bs, 40, nt * 16, 0, lane);
      acc[tt] = wmma_bf16(af, bfr, acc[tt]);
    }
    __syncthreads();
  }
  int half = lane >> 4, nr = lane & 15;
#pragma unroll
  for (int tt = 0; tt < 2; tt++) {
    int tile = t0 + tt;
    int mt = tile >> 2, nt = tile & 3;
#pragma unroll
    for (int r = 0; r < 8; r++) {
      int i = i0 + mt * 16 + r + 8 * half;
      int j = j0 + nt * 16 + nr;
      Out[(size_t)(i * 256 + j) * 128 + c] = acc[tt][r];
    }
  }
}

// ---------------------------------------------------------------------------
// Small elementwise / prep kernels
// ---------------------------------------------------------------------------
__global__ void trimul_ab_kernel(const __bf16* __restrict__ tBuf,
                                 const float* __restrict__ mask,
                                 __bf16* __restrict__ aOut, __bf16* __restrict__ bOut) {
  size_t idx = (size_t)blockIdx.x * 256 + threadIdx.x;  // over M*128
  size_t row = idx >> 7; int cc = (int)(idx & 127);
  float mk = mask[row];
  const __bf16* tr = tBuf + row * 512;
  float t0 = bf2f(tr[cc]), t1 = bf2f(tr[128 + cc]);
  float t2 = bf2f(tr[256 + cc]), t3 = bf2f(tr[384 + cc]);
  aOut[row * 128 + cc] = f2bf(t0 * sigmoidf(t1) * mk);
  bOut[row * 128 + cc] = f2bf(t2 * sigmoidf(t3) * mk);
}

__global__ void biasproj_kernel(const __bf16* __restrict__ xn,
                                const float* __restrict__ wb,
                                float* __restrict__ pbOut) {
  int idx = blockIdx.x * 256 + threadIdx.x;  // M*4
  int row = idx >> 2, h = idx & 3;
  const __bf16* xr = xn + (size_t)row * 128;
  float s = 0.0f;
  for (int c = 0; c < 128; c++) s += bf2f(xr[c]) * wb[c * 4 + h];
  pbOut[idx] = s;
}

__global__ void wprep_kernel(const float* __restrict__ W, __bf16* __restrict__ WT,
                             int K, int N, int ldWT, int rowOff) {
  int idx = blockIdx.x * 256 + threadIdx.x;
  if (idx >= K * N) return;
  int k = idx / N, n = idx % N;
  WT[(size_t)(rowOff + n) * ldWT + k] = f2bf(W[idx]);
}

__global__ void concat4_kernel(const float* __restrict__ a, const float* __restrict__ b,
                               const float* __restrict__ c, const float* __restrict__ d,
                               float* __restrict__ out) {
  int i = threadIdx.x;  // 512 threads
  float v;
  if (i < 128) v = a[i];
  else if (i < 256) v = b[i - 128];
  else if (i < 384) v = c[i - 256];
  else v = d[i - 384];
  out[i] = v;
}

// ---------------------------------------------------------------------------
// Host orchestration
// ---------------------------------------------------------------------------
struct AttnW { __bf16 *qkvgT, *woT; };
struct MulW  { __bf16 *wijsT, *wpT, *wsT; float* bias4; };

extern "C" void kernel_launch(void* const* d_in, const int* in_sizes, int n_in,
                              void* d_out, int out_size, void* d_ws, size_t ws_size,
                              hipStream_t stream) {
  (void)in_sizes; (void)n_in; (void)out_size; (void)ws_size;
  const int M = 65536;  // B*N*N
  const float* x2d  = (const float*)d_in[0];
  const float* mask = (const float*)d_in[1];
  auto P = [&](int i) { return (const float*)d_in[2 + i]; };

  size_t off = 0;
  auto alloc = [&](size_t bytes) -> void* {
    void* p = (char*)d_ws + off;
    off = (off + bytes + 255) & ~(size_t)255;
    return p;
  };
  __bf16* xn   = (__bf16*)alloc((size_t)M * 128 * 2);
  __bf16* big0 = (__bf16*)alloc((size_t)M * 512 * 2);  // also reused as f32 [M,128]
  __bf16* big1 = (__bf16*)alloc((size_t)M * 256 * 2);  // also reused as f32 [M,128]
  __bf16* nb   = (__bf16*)alloc((size_t)M * 128 * 2);
  float*  pbuf = (float*)alloc((size_t)M * 4 * 4);

  struct LayerW { AttnW tas, tae; MulW tmo, tmi; __bf16 *w1T, *w2T; } Lw[2];
  for (int l = 0; l < 2; l++) {
    Lw[l].tas.qkvgT = (__bf16*)alloc(512 * 128 * 2);
    Lw[l].tas.woT   = (__bf16*)alloc(128 * 128 * 2);
    Lw[l].tae.qkvgT = (__bf16*)alloc(512 * 128 * 2);
    Lw[l].tae.woT   = (__bf16*)alloc(128 * 128 * 2);
    Lw[l].tmo.wijsT = (__bf16*)alloc(512 * 128 * 2);
    Lw[l].tmo.wpT   = (__bf16*)alloc(128 * 128 * 2);
    Lw[l].tmo.wsT   = (__bf16*)alloc(128 * 128 * 2);
    Lw[l].tmo.bias4 = (float*)alloc(512 * 4);
    Lw[l].tmi.wijsT = (__bf16*)alloc(512 * 128 * 2);
    Lw[l].tmi.wpT   = (__bf16*)alloc(128 * 128 * 2);
    Lw[l].tmi.wsT   = (__bf16*)alloc(128 * 128 * 2);
    Lw[l].tmi.bias4 = (float*)alloc(512 * 4);
    Lw[l].w1T       = (__bf16*)alloc(512 * 128 * 2);
    Lw[l].w2T       = (__bf16*)alloc(128 * 512 * 2);
  }

  auto wprep = [&](const float* W, __bf16* WT, int K, int N, int ld, int rowOff) {
    int n = K * N;
    wprep_kernel<<<(n + 255) / 256, 256, 0, stream>>>(W, WT, K, N, ld, rowOff);
  };

  // Param index layout (jax pytree: dict keys sorted alphabetically):
  // per layer (58): pt[b1,b2,ln_b,ln_g,w1,w2] tae[bg,bo,ln_b,ln_g,wb,wg,wk,wo,wq,wv]
  //                 tas[...] tmi[bi,bis,bj,bjs,bp,bs,ln2_b,ln2_g,ln_b,ln_g,wi,wis,wj,wjs,wp,ws] tmo[...]
  // then final ln_b (116), ln_g (117).
  for (int l = 0; l < 2; l++) {
    int base = l * 58;
    int pt_ = base + 0, tae_ = base + 6, tas_ = base + 16, tmi_ = base + 26, tmo_ = base + 42;
    auto prepA = [&](int ai, AttnW& W) {
      wprep(P(ai + 8), W.qkvgT, 128, 128, 128, 0);    // wq
      wprep(P(ai + 6), W.qkvgT, 128, 128, 128, 128);  // wk
      wprep(P(ai + 9), W.qkvgT, 128, 128, 128, 256);  // wv
      wprep(P(ai + 5), W.qkvgT, 128, 128, 128, 384);  // wg
      wprep(P(ai + 7), W.woT,   128, 128, 128, 0);    // wo
    };
    auto prepM = [&](int mi, MulW& W) {
      wprep(P(mi + 10), W.wijsT, 128, 128, 128, 0);    // wi
      wprep(P(mi + 11), W.wijsT, 128, 128, 128, 128);  // wis
      wprep(P(mi + 12), W.wijsT, 128, 128, 128, 256);  // wj
      wprep(P(mi + 13), W.wijsT, 128, 128, 128, 384);  // wjs
      wprep(P(mi + 14), W.wpT, 128, 128, 128, 0);
      wprep(P(mi + 15), W.wsT, 128, 128, 128, 0);
      concat4_kernel<<<1, 512, 0, stream>>>(P(mi + 0), P(mi + 1), P(mi + 2), P(mi + 3), W.bias4);
    };
    prepA(tas_, Lw[l].tas);
    prepA(tae_, Lw[l].tae);
    prepM(tmo_, Lw[l].tmo);
    prepM(tmi_, Lw[l].tmi);
    wprep(P(pt_ + 4), Lw[l].w1T, 128, 512, 128, 0);  // w1 [128,512] -> [512][128]
    wprep(P(pt_ + 5), Lw[l].w2T, 512, 128, 512, 0);  // w2 [512,128] -> [128][512]
  }

  hipMemcpyAsync(d_out, x2d, (size_t)M * 128 * 4, hipMemcpyDeviceToDevice, stream);
  float* x = (float*)d_out;

  auto runAttn = [&](int ai, AttnW& W, bool starting) {
    ln_kernel<<<M / 8, 256, 0, stream>>>(x, P(ai + 3), P(ai + 2), xn, nullptr, nullptr, M);
    gemm_kernel<<<dim3(M / 128, 8), 256, 0, stream>>>(xn, 128, W.qkvgT, 128, nullptr, big0, 512,
                                                      nullptr, nullptr, nullptr, nullptr, GF_BF16OUT);
    biasproj_kernel<<<M * 4 / 256, 256, 0, stream>>>(xn, P(ai + 4), pbuf);
    if (starting)
      attn_kernel<<<dim3(256, 8), 256, 0, stream>>>(big0, pbuf, P(ai + 0), mask, big1,
                                                    256, 1, 1, 256, 256, 1);
    else
      attn_kernel<<<dim3(256, 8), 256, 0, stream>>>(big0, pbuf, P(ai + 0), mask, big1,
                                                    1, 256, 256, 1, 1, 256);
    gemm_kernel<<<dim3(M / 128, 2), 256, 0, stream>>>(big1, 128, W.woT, 128, x, nullptr, 128,
                                                      P(ai + 1), nullptr, mask, x, GF_MASKPRE);
  };
  auto runMul = [&](int mi, MulW& W, int iMul, int kMul) {
    ln_kernel<<<M / 8, 256, 0, stream>>>(x, P(mi + 9), P(mi + 8), xn, nullptr, nullptr, M);
    gemm_kernel<<<dim3(M / 128, 8), 256, 0, stream>>>(xn, 128, W.wijsT, 128, nullptr, big0, 512,
                                                      W.bias4, nullptr, nullptr, nullptr, GF_BF16OUT);
    __bf16* aB = big1;
    __bf16* bB = big1 + (size_t)M * 128;
    trimul_ab_kernel<<<M * 128 / 256, 256, 0, stream>>>(big0, mask, aB, bB);
    trimul_kernel<<<dim3(16, 128), 256, 0, stream>>>(aB, bB, (float*)big0, iMul, kMul);
    ln_kernel<<<M / 8, 256, 0, stream>>>((float*)big0, P(mi + 7), P(mi + 6), nb, nullptr, nullptr, M);
    gemm_kernel<<<dim3(M / 128, 2), 256, 0, stream>>>(xn, 128, W.wsT, 128, (float*)big1, nullptr, 128,
                                                      P(mi + 5), nullptr, nullptr, nullptr, 0);
    gemm_kernel<<<dim3(M / 128, 2), 256, 0, stream>>>(nb, 128, W.wpT, 128, x, nullptr, 128,
                                                      P(mi + 4), (float*)big1, mask, x, GF_MASKPRE);
  };
  auto runTrans = [&](int ti, __bf16* w1T, __bf16* w2T) {
    ln_kernel<<<M / 8, 256, 0, stream>>>(x, P(ti + 3), P(ti + 2), xn, nullptr, nullptr, M);
    gemm_kernel<<<dim3(M / 128, 8), 256, 0, stream>>>(xn, 128, w1T, 128, nullptr, big0, 512,
                                                      P(ti + 0), nullptr, nullptr, nullptr,
                                                      GF_RELU | GF_BF16OUT);
    gemm_kernel<<<dim3(M / 128, 2), 256, 0, stream>>>(big0, 512, w2T, 512, x, nullptr, 128,
                                                      P(ti + 1), nullptr, mask, x, GF_MASKPOST);
  };

  for (int l = 0; l < 2; l++) {
    int base = l * 58;
    int pt_ = base + 0, tae_ = base + 6, tas_ = base + 16, tmi_ = base + 26, tmo_ = base + 42;
    runAttn(tas_, Lw[l].tas, true);
    runAttn(tae_, Lw[l].tae, false);
    runMul(tmo_, Lw[l].tmo, 256, 1);   // bikc,bjkc->bijc
    runMul(tmi_, Lw[l].tmi, 1, 256);   // bkic,bkjc->bijc
    runTrans(pt_, Lw[l].w1T, Lw[l].w2T);
  }
  ln_kernel<<<M / 8, 256, 0, stream>>>(x, P(117), P(116), nullptr, x, mask, M);
}
